// DCLS_semi_DANNLayer_35399120454110
// MI455X (gfx1250) — matrix operand order repaired and checked by reading, not compiled, not run on hardware
//
#include <hip/hip_runtime.h>

// ---------------------------------------------------------------------------
// DCLS semi-DANN layer for MI455X (gfx1250), wave32 + v_wmma_f32_16x16x32_f16
// Pipeline:
//   1) build gaussian DCLS kernels -> f16, D padded 25->32 (K=32 per channel)
//   2) x (B,Ci,300) f32 -> f16 padded to T=320 (zero tail)
//   3) conv as GEMM  M=384(o) x N=17664(b,t) x K=700*32(i,d)  via WMMA f16
//      inh rows -> (b,t,i) f32 buffer, exc rows -> (o,b,t) f32 buffer
//   4) BN stats per channel (LDS reduction) -> scale/bias
//   5) LIF hard-reset scan over t (sequential, tiny) -> spikes f16 (tb, i)
//   6) GEMM2 256 x 17664 x 128 via WMMA f16, fused  out = excit + (-|W|)*spk
// ---------------------------------------------------------------------------

typedef _Float16 half8  __attribute__((ext_vector_type(8)));
typedef _Float16 half16 __attribute__((ext_vector_type(16)));
typedef float    float8 __attribute__((ext_vector_type(8)));

#define B_SZ   64
#define C_IN   700
#define T_IN   300
#define T_PAD  320
#define D_PAD  32
#define T_OUT  276
#define T_OPAD 288
#define N_INH  128
#define N_EXC  256
#define O_ALL  384
#define TBN    (T_OUT * B_SZ)   // 17664 columns of (t,b)

__device__ __forceinline__ half8 ldh8(const _Float16* p) {
    half8 r; __builtin_memcpy(&r, p, 16); return r;
}
__device__ __forceinline__ half16 ldh16u(const _Float16* p) {
    half16 r; __builtin_memcpy(&r, p, 32); return r;
}
__device__ __forceinline__ half16 joinA(half8 lo, half8 hi) {
    return __builtin_shufflevector(lo, hi, 0,1,2,3,4,5,6,7,8,9,10,11,12,13,14,15);
}
#define WMMA16(A, Bm, C) \
    __builtin_amdgcn_wmma_f32_16x16x32_f16(false, (A), false, (Bm), (short)0, (C), false, false)

// --------------------------- 1) DCLS kernel build ---------------------------
__global__ void build_kernel_f16(const float* __restrict__ W,
                                 const float* __restrict__ P,
                                 const float* __restrict__ S,
                                 _Float16* __restrict__ kf16,
                                 int rows, int rowOffset) {
    int idx = blockIdx.x * blockDim.x + threadIdx.x;     // over rows * C_IN
    if (idx >= rows * C_IN) return;
    float w   = fabsf(W[idx]);
    float pc  = fminf(fmaxf(P[idx], -12.f), 12.f) + 12.f;
    float isg = 1.f / (fabsf(S[idx]) + 0.27f);
    float g[25]; float sum = 0.f;
#pragma unroll
    for (int d = 0; d < 25; ++d) {
        float z = ((float)d - pc) * isg;
        float e = __expf(-0.5f * z * z);
        g[d] = e; sum += e;
    }
    float sc = w / (sum + 1e-7f);
    _Float16* dst = kf16 + ((size_t)rowOffset * C_IN + idx) * D_PAD;
#pragma unroll
    for (int d = 0; d < 25; ++d) dst[d] = (_Float16)(g[d] * sc);
#pragma unroll
    for (int d = 25; d < D_PAD; ++d) dst[d] = (_Float16)0.f;
}

// --------------------------- 2) x -> f16, pad T ---------------------------
__global__ void x_to_f16(const float* __restrict__ x, _Float16* __restrict__ xf) {
    size_t idx = (size_t)blockIdx.x * blockDim.x + threadIdx.x;
    if (idx >= (size_t)B_SZ * C_IN * T_PAD) return;
    int t = (int)(idx % T_PAD);
    size_t bi = idx / T_PAD;
    xf[idx] = (t < T_IN) ? (_Float16)x[bi * T_IN + t] : (_Float16)0.f;
}

// --------------------------- w2 = -|w_exc_inh| -> f16 ---------------------------
__global__ void w2_to_f16(const float* __restrict__ w, _Float16* __restrict__ wf) {
    int idx = blockIdx.x * blockDim.x + threadIdx.x;
    if (idx >= N_EXC * N_INH) return;
    wf[idx] = (_Float16)(-fabsf(w[idx]));
}

// --------------------------- 3) conv as WMMA GEMM ---------------------------
// grid = (3 t-tiles of 96, 3 o-tiles of 128, 64 b); block = 256 (8 waves)
__global__ __launch_bounds__(256) void conv_wmma(
        const _Float16* __restrict__ kf16, const _Float16* __restrict__ xf16,
        float* __restrict__ inh_bt, float* __restrict__ excout) {
    const int t0   = blockIdx.x * 96;
    const int b    = blockIdx.z;
    const int wave = threadIdx.x >> 5;
    const int lane = threadIdx.x & 31;
    const int m    = lane & 15;
    const int hi   = lane >> 4;
    const int o0   = blockIdx.y * 128 + wave * 16;

    float8 c[6] = {};
    const _Float16* arow = kf16 + (size_t)(o0 + m) * C_IN * D_PAD + hi * 16;
    const _Float16* xrow = xf16 + (size_t)b * C_IN * T_PAD + t0 + m + hi * 16;

    for (int ic = 0; ic < C_IN; ++ic) {
        half16 a = ldh16u(arow);
#pragma unroll
        for (int tt = 0; tt < 6; ++tt) {
            half16 bf = ldh16u(xrow + tt * 16);      // contiguous t => implicit im2col
            c[tt] = WMMA16(a, bf, c[tt]);
        }
        arow += D_PAD;
        xrow += T_PAD;
    }

    const int obase = o0 + hi * 8;
#pragma unroll
    for (int tt = 0; tt < 6; ++tt) {
        int t = t0 + tt * 16 + m;
#pragma unroll
        for (int r = 0; r < 8; ++r) {
            int o = obase + r;
            float v = c[tt][r];
            if (o0 < N_INH)
                inh_bt[((size_t)b * T_OPAD + t) * N_INH + o] = v;
            else
                excout[((size_t)(o - N_INH) * B_SZ + b) * T_OPAD + t] = v;
        }
    }
}

// --------------------------- 4) BN stats (per channel) ---------------------------
__global__ void bn_stats(const float* __restrict__ inh_bt,
                         const float* __restrict__ gamma,
                         const float* __restrict__ beta,
                         float* __restrict__ bnp) {
    __shared__ float sh[512];
    const int i = blockIdx.x;          // channel 0..127
    const int tid = threadIdx.x;       // 256
    float s = 0.f, s2 = 0.f;
    for (int tb = tid; tb < TBN; tb += 256) {
        int t = tb >> 6, b = tb & 63;
        float v = inh_bt[((size_t)b * T_OPAD + t) * N_INH + i];
        s += v; s2 += v * v;
    }
    sh[tid] = s; sh[256 + tid] = s2;
    __syncthreads();
    for (int off = 128; off > 0; off >>= 1) {
        if (tid < off) { sh[tid] += sh[tid + off]; sh[256 + tid] += sh[256 + tid + off]; }
        __syncthreads();
    }
    if (tid == 0) {
        float mean = sh[0] * (1.f / TBN);
        float var  = sh[256] * (1.f / TBN) - mean * mean;
        float rstd = rsqrtf(var + 1e-5f);
        float sc   = gamma[i] * rstd;
        bnp[i] = sc;
        bnp[N_INH + i] = beta[i] - mean * sc;
    }
}

// --------------------------- 5) LIF hard-reset scan ---------------------------
// tau=2 -> v = 0.5*v + x ; spike = (v >= 1) ; v *= (1 - spike)
__global__ void lif_scan(const float* __restrict__ inh_bt,
                         const float* __restrict__ bnp,
                         _Float16* __restrict__ spikes) {
    const int b = blockIdx.x;          // 0..63
    const int i = threadIdx.x;         // 0..127
    const float sc = bnp[i], bi = bnp[N_INH + i];
    float v = 0.f;
    for (int t = 0; t < T_OUT; ++t) {
        float x = inh_bt[((size_t)b * T_OPAD + t) * N_INH + i] * sc + bi;
        v = 0.5f * v + x;
        float s = (v >= 1.0f) ? 1.0f : 0.0f;
        v = v * (1.0f - s);
        spikes[((size_t)t * B_SZ + b) * N_INH + i] = (_Float16)s;
    }
}

// --------------------------- 6) GEMM2 + fused final add ---------------------------
// grid = (184 n-tiles of 96 tb-columns, 2 o-tiles of 128); block = 256 (8 waves)
__global__ __launch_bounds__(256) void gemm2_wmma(
        const _Float16* __restrict__ w2, const _Float16* __restrict__ spikes,
        const float* __restrict__ excout, float* __restrict__ out) {
    const int ntile = blockIdx.x;
    const int wave  = threadIdx.x >> 5;
    const int lane  = threadIdx.x & 31;
    const int m     = lane & 15;
    const int hi    = lane >> 4;
    const int o0    = blockIdx.y * 128 + wave * 16;
    const int tb0   = ntile * 96;

    float8 c[6] = {};
#pragma unroll
    for (int kk = 0; kk < 4; ++kk) {                 // K = 128 = 4 x 32
        const _Float16* ar = w2 + (size_t)(o0 + m) * N_INH + kk * 32 + hi * 16;
        half16 a = ldh16u(ar);
#pragma unroll
        for (int tt = 0; tt < 6; ++tt) {
            const _Float16* br = spikes + (size_t)(tb0 + tt * 16 + m) * N_INH
                                        + kk * 32 + hi * 16;
            half16 bf = ldh16u(br);
            c[tt] = WMMA16(a, bf, c[tt]);
        }
    }

    const int obase = o0 + hi * 8;
#pragma unroll
    for (int tt = 0; tt < 6; ++tt) {
        int tb = tb0 + tt * 16 + m;
        int t = tb >> 6, b = tb & 63;
#pragma unroll
        for (int r = 0; r < 8; ++r) {
            int o = obase + r;
            float e = excout[((size_t)o * B_SZ + b) * T_OPAD + t];
            out[((size_t)b * N_EXC + o) * T_OUT + t] = e + c[tt][r];
        }
    }
}

// --------------------------- launch ---------------------------
extern "C" void kernel_launch(void* const* d_in, const int* in_sizes, int n_in,
                              void* d_out, int out_size, void* d_ws, size_t ws_size,
                              hipStream_t stream) {
    const float* x        = (const float*)d_in[0];
    const float* W_inh    = (const float*)d_in[1];
    const float* P_inh    = (const float*)d_in[2];
    const float* SIG_inh  = (const float*)d_in[3];
    const float* W_exc    = (const float*)d_in[4];
    const float* P_exc    = (const float*)d_in[5];
    const float* SIG_exc  = (const float*)d_in[6];
    const float* w_ei     = (const float*)d_in[7];
    const float* bn_gamma = (const float*)d_in[8];
    const float* bn_beta  = (const float*)d_in[9];
    float* out = (float*)d_out;

    char* p = (char*)d_ws;
    auto carve = [&](size_t bytes) -> char* {
        char* r = p; p += (bytes + 255) & ~(size_t)255; return r;
    };
    _Float16* kf16   = (_Float16*)carve((size_t)O_ALL * C_IN * D_PAD * 2);  // 17.2 MB
    _Float16* xf16   = (_Float16*)carve((size_t)B_SZ * C_IN * T_PAD * 2);   // 28.7 MB
    _Float16* w2f16  = (_Float16*)carve((size_t)N_EXC * N_INH * 2);
    float*    inh_bt = (float*)carve((size_t)B_SZ * T_OPAD * N_INH * 4);    //  9.4 MB
    float*    excout = (float*)carve((size_t)N_EXC * B_SZ * T_OPAD * 4);    // 18.9 MB
    _Float16* spikes = (_Float16*)carve((size_t)TBN * N_INH * 2);           //  4.5 MB
    float*    bnp    = (float*)carve((size_t)2 * N_INH * 4);
    (void)ws_size; (void)in_sizes; (void)n_in; (void)out_size;

    // 1) kernels
    build_kernel_f16<<<(N_INH * C_IN + 255) / 256, 256, 0, stream>>>(
        W_inh, P_inh, SIG_inh, kf16, N_INH, 0);
    build_kernel_f16<<<(N_EXC * C_IN + 255) / 256, 256, 0, stream>>>(
        W_exc, P_exc, SIG_exc, kf16, N_EXC, N_INH);
    // 2) x -> f16
    {
        size_t n = (size_t)B_SZ * C_IN * T_PAD;
        x_to_f16<<<(unsigned)((n + 255) / 256), 256, 0, stream>>>(x, xf16);
    }
    // w2
    w2_to_f16<<<(N_EXC * N_INH + 255) / 256, 256, 0, stream>>>(w_ei, w2f16);
    // 3) conv GEMM  (3 t-tiles x 3 o-tiles x 64 b)
    conv_wmma<<<dim3(T_OPAD / 96, O_ALL / 128, B_SZ), 256, 0, stream>>>(
        kf16, xf16, inh_bt, excout);
    // 4) BN stats
    bn_stats<<<N_INH, 256, 0, stream>>>(inh_bt, bn_gamma, bn_beta, bnp);
    // 5) LIF scan
    lif_scan<<<B_SZ, N_INH, 0, stream>>>(inh_bt, bnp, spikes);
    // 6) GEMM2 + final add  (184 n-tiles x 2 o-tiles)
    gemm2_wmma<<<dim3(TBN / 96, N_EXC / 128), 256, 0, stream>>>(
        w2f16, spikes, excout, out);
}
